// ModelClass_48644799595231
// MI455X (gfx1250) — matrix operand order repaired and checked by reading, not compiled, not run on hardware
//
#include <hip/hip_runtime.h>
#include <hip/hip_bf16.h>
#include <stdint.h>

// Problem constants (match reference)
#define N_NODESC 262144
#define N_EDGESC 4194304
#define N_GRAPHSC 128
#define DYNF 32
#define NSTATIC 4
#define HLF 6
#define NALL 42          // DYN + STATIC + HL
#define APAD 44          // 42 padded to multiple of 4 floats (16B)
#define HIDN 512
#define K1PAD 64         // 42 padded to multiple of 32 for WMMA K
#define NPROPC 3

typedef __bf16 v16bf __attribute__((ext_vector_type(16)));
typedef float  v8f   __attribute__((ext_vector_type(8)));

// Branchless fp32 -> bf16 (RNE, NaN-safe): lowers to v_cndmask, no EXEC divergence.
__device__ __forceinline__ unsigned short f2bf(float f) {
  union { float f; unsigned u; } c; c.f = f;
  unsigned u = c.u;
  unsigned rn = (u + 0x7fffu + ((u >> 16) & 1u)) >> 16;   // round-to-nearest-even
  unsigned nn = (u >> 16) | 0x40u;                        // quieted NaN
  bool isnan = (u & 0x7fffffffu) > 0x7f800000u;
  return (unsigned short)(isnan ? nn : rn);
}

// ---------------- weight transpose + fp32 -> bf16 (once per launch) ---------
// W is [K, ncol] row-major fp32 (jax default). WT is [ncol, Kpad] bf16.
__global__ void k_wt(const float* __restrict__ W, unsigned short* __restrict__ WT,
                     int K, int Kpad, int ncol) {
  int i = blockIdx.x * blockDim.x + threadIdx.x;
  if (i >= ncol * Kpad) return;
  int o = i / Kpad, k = i % Kpad;
  WT[i] = (k < K) ? f2bf(W[(size_t)k * ncol + o]) : (unsigned short)0;
}

// ---------------- elementwise helpers ---------------------------------------
__global__ void k_init_h(const float* __restrict__ x, float* __restrict__ h) {
  int i = blockIdx.x * blockDim.x + threadIdx.x;     // over N*32
  int node = i >> 5, c = i & 31;
  h[i] = (c == 0) ? x[node] : 0.f;
}

// a44 = [h | static | hlvs[batch]] padded to 44; also zero agg44
__global__ void k_build_a(const float* __restrict__ h, const float* __restrict__ stat,
                          const float* __restrict__ hlvs, const int* __restrict__ batch,
                          float* __restrict__ a44, float* __restrict__ agg44) {
  int i = blockIdx.x * blockDim.x + threadIdx.x;     // over N*44
  int node = i / APAD, c = i - node * APAD;
  float v = 0.f;
  if (c < DYNF)                 v = h[node * DYNF + c];
  else if (c < DYNF + NSTATIC)  v = stat[node * NSTATIC + (c - DYNF)];
  else if (c < NALL)            v = hlvs[batch[node] * HLF + (c - DYNF - NSTATIC)];
  a44[i] = v;
  agg44[i] = 0.f;
}

// agg44[dst] += a44[src]  (GIN sum aggregation) — one wave per edge:
// coalesced row gather + 42 lane-parallel float atomics (2 atomic instructions).
__global__ __launch_bounds__(256) void k_edges(const int* __restrict__ src,
                                               const int* __restrict__ dst,
                                               const float* __restrict__ a44,
                                               float* __restrict__ agg44) {
  const int e    = blockIdx.x * 8 + (threadIdx.x >> 5);   // wave id = edge id
  const int lane = threadIdx.x & 31;
  const int s = src[e];
  const int d = dst[e];
  const float* ap = a44   + (size_t)s * APAD;
  float*       gp = agg44 + (size_t)d * APAD;
  float v0 = ap[lane];                       // cols 0..31, coalesced
  atomicAdd(gp + lane, v0);
  if (lane < NALL - 32) {                    // cols 32..41
    float v1 = ap[32 + lane];
    atomicAdd(gp + 32 + lane, v1);
  }
}

// zb = bf16( (1+eps)*a + agg ), [N, 64] zero-padded
__global__ void k_z(const float* __restrict__ a44, const float* __restrict__ agg44,
                    const float* __restrict__ epsp, unsigned short* __restrict__ zb) {
  int i = blockIdx.x * blockDim.x + threadIdx.x;     // over N*64
  int node = i >> 6, c = i & 63;
  float v = 0.f;
  if (c < NALL) {
    float e1 = 1.f + *epsp;
    v = e1 * a44[(size_t)node * APAD + c] + agg44[(size_t)node * APAD + c];
  }
  zb[i] = f2bf(v);
}

// zb = bf16([h | static | hlvs[batch]]), [N, 64] zero-padded (node MLP input)
__global__ void k_nodein(const float* __restrict__ h, const float* __restrict__ stat,
                         const float* __restrict__ hlvs, const int* __restrict__ batch,
                         unsigned short* __restrict__ zb) {
  int i = blockIdx.x * blockDim.x + threadIdx.x;     // over N*64
  int node = i >> 6, c = i & 63;
  float v = 0.f;
  if (c < DYNF)                 v = h[node * DYNF + c];
  else if (c < DYNF + NSTATIC)  v = stat[node * NSTATIC + (c - DYNF)];
  else if (c < NALL)            v = hlvs[batch[node] * HLF + (c - DYNF - NSTATIC)];
  zb[i] = f2bf(v);
}

// ---------------- bf16 WMMA GEMM: out = relu(A[N,KDIM] * WT^T + bias) -------
// A: [N, KDIM] bf16 row-major.  WT: [ncol, KDIM] bf16 (transposed weights).
// Block: 256 threads = 8 waves; wave w -> rows [by*128 + 16w, +16), COLT 16-col tiles.
// COLT*16-column weight slab staged in LDS with 16B xor swizzle (bank-conflict free).
// N multiple of 128 and ncol multiple of COLT*16 -> no divergence, EXEC all ones.
template <int KDIM, int COLT, bool OUT_BF16>
__global__ __launch_bounds__(256) void k_gemm(const unsigned short* __restrict__ A,
                                              const unsigned short* __restrict__ WT,
                                              const float* __restrict__ bias,
                                              void* __restrict__ outp, int outStride) {
  constexpr int COLS  = COLT * 16;
  constexpr int GPC   = KDIM / 8;                       // 16B groups per column
  constexpr int GMASK = (GPC - 1) < 15 ? (GPC - 1) : 15;
  __shared__ unsigned short lds[COLS * KDIM];

  const int tid = threadIdx.x;
  const int colBase = blockIdx.x * COLS;

  // stage weight slab -> LDS (swizzled at 16B granularity)
  for (int i = tid; i < COLS * GPC; i += 256) {
    int col = i / GPC, g = i - col * GPC;
    uint4 v = *(const uint4*)(WT + (size_t)(colBase + col) * KDIM + (size_t)g * 8);
    *(uint4*)(&lds[col * KDIM + ((g ^ (col & GMASK)) << 3)]) = v;
  }
  __syncthreads();

  const int wave = tid >> 5;
  const int lane = tid & 31;
  const int half = lane >> 4;        // 0: K lo-half pattern, 1: hi-half pattern
  const int m    = lane & 15;        // row (A) / column (B,C,D) within tile
  const int row  = blockIdx.y * 128 + wave * 16 + m;

  union FragA { v16bf v; uint4 q[2]; } fa;
  union FragB { v16bf v; uint4 q[2]; } fb;
  const v8f vzero = {0.f, 0.f, 0.f, 0.f, 0.f, 0.f, 0.f, 0.f};
  v8f acc[COLT];
#pragma unroll
  for (int t = 0; t < COLT; ++t) acc[t] = vzero;

  const unsigned short* Arow = A + (size_t)row * KDIM;
  for (int k0 = 0; k0 < KDIM; k0 += 32) {
    // A 16x32 bf16 fragment: lane<16 holds K {0..7,16..23}, lane>=16 holds {8..15,24..31}
    int kA = k0 + half * 8;
    fa.q[0] = *(const uint4*)(Arow + kA);
    fa.q[1] = *(const uint4*)(Arow + kA + 16);
    int gB = (k0 + half * 8) >> 3;   // B: lane<16 K {0..7,16..23}, lane>=16 K {8..15,24..31} (col = m)
#pragma unroll
    for (int t = 0; t < COLT; ++t) {
      int col = t * 16 + m;
      const unsigned short* L = lds + col * KDIM;
      fb.q[0] = *(const uint4*)(L + (((gB    ) ^ (col & GMASK)) << 3));
      fb.q[1] = *(const uint4*)(L + (((gB + 2) ^ (col & GMASK)) << 3));
      acc[t] = __builtin_amdgcn_wmma_f32_16x16x32_bf16(
          false, fa.v, false, fb.v, (short)0, acc[t], false, false);
    }
  }

  // epilogue: bias + ReLU, scatter per C/D layout (lane half -> rows +0 / +8).
  // Each half-wave writes 16 consecutive bf16 per row -> 32B coalesced chunks.
  const int rbase = blockIdx.y * 128 + wave * 16 + half * 8;
#pragma unroll
  for (int t = 0; t < COLT; ++t) {
    int c = colBase + t * 16 + m;
    float bv = bias[c];
#pragma unroll
    for (int v = 0; v < 8; ++v) {
      float val = acc[t][v] + bv;
      val = val > 0.f ? val : 0.f;
      size_t off = (size_t)(rbase + v) * (size_t)outStride + c;
      if (OUT_BF16) ((unsigned short*)outp)[off] = f2bf(val);
      else          ((float*)outp)[off] = val;
    }
  }
}

// ---------------- pooling + tiny final MLP ----------------------------------
__global__ void k_pool(const float* __restrict__ h, const int* __restrict__ batch,
                       float* __restrict__ pooled) {
  int i = blockIdx.x * blockDim.x + threadIdx.x;     // over N*32
  int node = i >> 5, c = i & 31;
  atomicAdd(&pooled[batch[node] * DYNF + c], h[i]);
}

__global__ __launch_bounds__(256) void k_final(
    const float* __restrict__ hlvs, const float* __restrict__ pooled,
    const float* __restrict__ W1, const float* __restrict__ b1,
    const float* __restrict__ W2, const float* __restrict__ b2,
    const float* __restrict__ W3, const float* __restrict__ b3,
    const float* __restrict__ W4, const float* __restrict__ b4,
    float* __restrict__ out) {
  __shared__ float buf0[HIDN];
  __shared__ float buf1[HIDN];
  __shared__ float red[256];
  const int g = blockIdx.x, tid = threadIdx.x;
  const int KIN = HLF + DYNF;  // 38

  for (int i = tid; i < HIDN; i += 256) buf0[i] = 0.f;
  if (tid < HLF)            buf0[tid] = hlvs[g * HLF + tid];
  else if (tid < KIN)       buf0[tid] = pooled[g * DYNF + (tid - HLF)];
  __syncthreads();

  for (int o = tid; o < HIDN; o += 256) {
    float s = b1[o];
    for (int k = 0; k < KIN; ++k) s += buf0[k] * W1[(size_t)k * HIDN + o];
    buf1[o] = s > 0.f ? s : 0.f;
  }
  __syncthreads();
  for (int o = tid; o < HIDN; o += 256) {
    float s = b2[o];
    for (int k = 0; k < HIDN; ++k) s += buf1[k] * W2[(size_t)k * HIDN + o];
    buf0[o] = s > 0.f ? s : 0.f;
  }
  __syncthreads();
  for (int o = tid; o < HIDN; o += 256) {
    float s = b3[o];
    for (int k = 0; k < HIDN; ++k) s += buf0[k] * W3[(size_t)k * HIDN + o];
    buf1[o] = s > 0.f ? s : 0.f;
  }
  __syncthreads();
  float p = 0.f;
  for (int k = tid; k < HIDN; k += 256) p += buf1[k] * W4[k];
  red[tid] = p;
  __syncthreads();
  for (int s2 = 128; s2 > 0; s2 >>= 1) {
    if (tid < s2) red[tid] += red[tid + s2];
    __syncthreads();
  }
  if (tid == 0) {
    float r = red[0] + b4[0];
    out[g] = r > 0.f ? r : 0.f;
  }
}

// ---------------- host orchestration ----------------------------------------
extern "C" void kernel_launch(void* const* d_in, const int* in_sizes, int n_in,
                              void* d_out, int out_size, void* d_ws, size_t ws_size,
                              hipStream_t stream) {
  (void)in_sizes; (void)n_in; (void)out_size; (void)ws_size;
  // Input order (pytree-flattened setup_inputs dict):
  const float* x     = (const float*)d_in[0];
  const float* stat  = (const float*)d_in[1];
  const float* hlvs  = (const float*)d_in[2];
  const int*   batch = (const int*)d_in[3];
  const int*   eidx  = (const int*)d_in[4];
  const int*   esrc  = eidx;
  const int*   edst  = eidx + N_EDGESC;
  const float* cW[4] = {(const float*)d_in[6],  (const float*)d_in[8],
                        (const float*)d_in[10], (const float*)d_in[12]};
  const float* cb[4] = {(const float*)d_in[7],  (const float*)d_in[9],
                        (const float*)d_in[11], (const float*)d_in[13]};
  const float* eps   = (const float*)d_in[14];
  const float* nW[4] = {(const float*)d_in[15], (const float*)d_in[17],
                        (const float*)d_in[19], (const float*)d_in[21]};
  const float* nb[4] = {(const float*)d_in[16], (const float*)d_in[18],
                        (const float*)d_in[20], (const float*)d_in[22]};
  const float* hW[4] = {(const float*)d_in[23], (const float*)d_in[25],
                        (const float*)d_in[27], (const float*)d_in[29]};
  const float* hb[4] = {(const float*)d_in[24], (const float*)d_in[26],
                        (const float*)d_in[28], (const float*)d_in[30]};
  float* out = (float*)d_out;

  // workspace carve (~700 MB total)
  char* p = (char*)d_ws;
  auto carve = [&](size_t bytes) -> void* {
    void* r = (void*)p;
    p += (bytes + 255) & ~(size_t)255;
    return r;
  };
  float* h     = (float*)carve((size_t)N_NODESC * DYNF * 4);
  float* a44   = (float*)carve((size_t)N_NODESC * APAD * 4);
  float* agg44 = (float*)carve((size_t)N_NODESC * APAD * 4);
  unsigned short* zb = (unsigned short*)carve((size_t)N_NODESC * K1PAD * 2);
  unsigned short* y1 = (unsigned short*)carve((size_t)N_NODESC * HIDN * 2);
  unsigned short* y2 = (unsigned short*)carve((size_t)N_NODESC * HIDN * 2);
  float* pooled = (float*)carve((size_t)N_GRAPHSC * DYNF * 4);

  const int Ks[4]    = {NALL, HIDN, HIDN, HIDN};
  const int Kpads[4] = {K1PAD, HIDN, HIDN, HIDN};
  const int ncols[4] = {HIDN, HIDN, HIDN, DYNF};
  unsigned short* cWT[4];
  unsigned short* nWT[4];
  for (int l = 0; l < 4; ++l) {
    cWT[l] = (unsigned short*)carve((size_t)ncols[l] * Kpads[l] * 2);
    nWT[l] = (unsigned short*)carve((size_t)ncols[l] * Kpads[l] * 2);
  }

  // convert + transpose weights to bf16 (graph-capture safe, deterministic)
  for (int l = 0; l < 4; ++l) {
    int tot = ncols[l] * Kpads[l];
    k_wt<<<(tot + 255) / 256, 256, 0, stream>>>(cW[l], cWT[l], Ks[l], Kpads[l], ncols[l]);
    k_wt<<<(tot + 255) / 256, 256, 0, stream>>>(nW[l], nWT[l], Ks[l], Kpads[l], ncols[l]);
  }

  k_init_h<<<N_NODESC * DYNF / 256, 256, 0, stream>>>(x, h);

  const dim3 gBig(HIDN / 64, N_NODESC / 128);   // (col groups fastest, row blocks)
  const dim3 gLast(DYNF / 32, N_NODESC / 128);

  for (int it = 0; it < NPROPC; ++it) {
    k_build_a<<<N_NODESC * APAD / 256, 256, 0, stream>>>(h, stat, hlvs, batch, a44, agg44);
    k_edges<<<N_EDGESC / 8, 256, 0, stream>>>(esrc, edst, a44, agg44);
    k_z<<<N_NODESC * K1PAD / 256, 256, 0, stream>>>(a44, agg44, eps, zb);
    // conv MLP: 42->512->512->512->32, all layers ReLU
    k_gemm<K1PAD, 4, true ><<<gBig,  256, 0, stream>>>(zb, cWT[0], cb[0], y1, HIDN);
    k_gemm<HIDN,  4, true ><<<gBig,  256, 0, stream>>>(y1, cWT[1], cb[1], y2, HIDN);
    k_gemm<HIDN,  4, true ><<<gBig,  256, 0, stream>>>(y2, cWT[2], cb[2], y1, HIDN);
    k_gemm<HIDN,  2, false><<<gLast, 256, 0, stream>>>(y1, cWT[3], cb[3], h,  DYNF);
    // node MLP on [h | static | hlv]
    k_nodein<<<N_NODESC * K1PAD / 256, 256, 0, stream>>>(h, stat, hlvs, batch, zb);
    k_gemm<K1PAD, 4, true ><<<gBig,  256, 0, stream>>>(zb, nWT[0], nb[0], y1, HIDN);
    k_gemm<HIDN,  4, true ><<<gBig,  256, 0, stream>>>(y1, nWT[1], nb[1], y2, HIDN);
    k_gemm<HIDN,  4, true ><<<gBig,  256, 0, stream>>>(y2, nWT[2], nb[2], y1, HIDN);
    k_gemm<HIDN,  2, false><<<gLast, 256, 0, stream>>>(y1, nWT[3], nb[3], h,  DYNF);
  }

  hipMemsetAsync(pooled, 0, (size_t)N_GRAPHSC * DYNF * 4, stream);
  k_pool<<<N_NODESC * DYNF / 256, 256, 0, stream>>>(h, batch, pooled);
  k_final<<<N_GRAPHSC, 256, 0, stream>>>(hlvs, pooled, hW[0], hb[0], hW[1], hb[1],
                                         hW[2], hb[2], hW[3], hb[3], out);
}
